// HRM_67628555043153
// MI455X (gfx1250) — matrix operand order repaired and checked by reading, not compiled, not run on hardware
//
#include <hip/hip_runtime.h>
#include <hip/hip_bf16.h>

// ---------------------------------------------------------------------------
// HRM forward pass for MI455X (gfx1250, wave32, WMMA).
//   D=128 DL=160 DH=192 NC=10 NT=1000 ML=400 B=1024 K=2 L=400
// Exact-algebra simplifications:
//   * support encoding  -> color histogram @ tok + pos_mean
//   * decode fc1        -> tokW[q] + posW[l] + pb[b]  (fc1 linear in concat)
// All linears but decode fc2 (scalar f32) run on v_wmma_f32_16x16x32_f16 (f16 in / f32 accumulate).
// Weights are pre-packed once to f16 in the GEMM's contiguous-k wave32 B-fragment layout;
// the GEMM helper is templated on (K,N) and fully unrolled, k-loop outermost,
// so A converts once per k-step and B loads pipeline ahead of the WMMA chain.
// ---------------------------------------------------------------------------

typedef __attribute__((ext_vector_type(16))) _Float16 v16h;
typedef __attribute__((ext_vector_type(8)))  float    v8f;
typedef __attribute__((ext_vector_type(4)))  float    v4f;

#define BD 128              // threads per chain block (4 waves)
#define NB 1024             // batch
#define LQ 400              // sequence length
#define DD 128

__device__ __forceinline__ float gelu_f(float x) {
    return 0.5f * x * (1.0f + erff(x * 0.70710678118654752440f));
}

// ---------------------------------------------------------------------------
// Weight packing: dst[((k/32)*N + n)*32 + ((k%32)/16)*16 + (k%16)]
//               = (f16) W[k*N + n]
// For each 32-row k-tile, lane (n, half) finds its 16 B-fragment halves
// contiguous (32 bytes) -> two b128 loads in the GEMM hot loop.
// ---------------------------------------------------------------------------
__global__ __launch_bounds__(256) void packw_kernel(
    const float* __restrict__ W, _Float16* __restrict__ dst, int K, int N)
{
    const int total = K * N;
    for (int o = blockIdx.x * 256 + threadIdx.x; o < total;
         o += gridDim.x * 256) {
        int e    = o & 15;
        int half = (o >> 4) & 1;
        int rest = o >> 5;
        int n    = rest % N;
        int kt   = rest / N;
        int k    = kt * 32 + half * 16 + e;
        dst[o] = (_Float16)W[(size_t)k * N + n];
    }
}

// ---------------------------------------------------------------------------
// OUT[16][N] (LDS, stride outStride) (+)= ACT[16][K] @ W[K][N] (+ bias)
// ACT f32 in LDS; W pre-packed f16.  Executed by all 4 waves of the block.
// Fully unrolled; k outermost so the A fragment is built once per k-step and
// reused across this wave's NT output tiles.
// Fragment layouts used here (wave32; A/B in contiguous k, not ISA table order):
//   A 16x32 f16 : lane M = lane&15; lanes<16 hold K0..15, lanes>=16
//                 hold K16..31
//   B 32x16 f16 : lane N = lane&15; lanes<16 hold K0..15, lanes>=16 K16..31
//   C/D 16x16 f32: vgpr j -> M = j + 8*(lane>=16), N = lane&15
// ---------------------------------------------------------------------------
template <int K, int N>
__device__ __forceinline__ void block_gemm16(
    const float* act, int actStride,
    const _Float16* __restrict__ Wp,
    const float* __restrict__ bias,
    float* outp, int outStride,
    int accumulate)
{
    constexpr int NTILES = N / 16;
    constexpr int NT     = (NTILES + 3) / 4;           // tiles per wave
    constexpr bool FULL  = (NTILES % 4) == 0;
    __syncthreads();
    const int lane = threadIdx.x & 31;
    const int wave = threadIdx.x >> 5;
    const int half = lane >> 4;
    const int l15  = lane & 15;
    const float* abase = act + l15 * actStride + half * 16;
    const _Float16* wbase = Wp + (size_t)l15 * 32 + half * 16;

    v8f acc[NT];
#pragma unroll
    for (int t = 0; t < NT; ++t) acc[t] = (v8f){};

#pragma unroll
    for (int kt = 0; kt < K / 32; ++kt) {
        float av[16];
        *(v4f*)(av + 0)  = *(const v4f*)(abase + kt * 32);
        *(v4f*)(av + 4)  = *(const v4f*)(abase + kt * 32 + 4);
        *(v4f*)(av + 8)  = *(const v4f*)(abase + kt * 32 + 8);
        *(v4f*)(av + 12) = *(const v4f*)(abase + kt * 32 + 12);
        v16h a;
#pragma unroll
        for (int e = 0; e < 16; ++e) a[e] = (_Float16)av[e];
#pragma unroll
        for (int t = 0; t < NT; ++t) {
            const int tile = wave + t * 4;
            if (FULL || tile < NTILES) {
                v16h b = *(const v16h*)(wbase + ((size_t)kt * N + tile * 16) * 32);
                acc[t] = __builtin_amdgcn_wmma_f32_16x16x32_f16(
                    false, a, false, b, (short)0, acc[t], false, false);
            }
        }
    }

#pragma unroll
    for (int t = 0; t < NT; ++t) {
        const int tile = wave + t * 4;
        if (FULL || tile < NTILES) {
            const int n = tile * 16 + l15;
#pragma unroll
            for (int j = 0; j < 8; ++j) {
                int m = j + half * 8;
                float v = acc[t][j];
                if (bias)       v += bias[n];
                if (accumulate) v += outp[m * outStride + n];
                outp[m * outStride + n] = v;
            }
        }
    }
    __syncthreads();
}

// ---------------------------------------------------------------------------
// Per-batch color histograms of support_inp/support_out, normalized 1/(K*L).
// cnt layout: [b][0..9]=inp, [b][16..25]=out (stride 32)
// ---------------------------------------------------------------------------
__global__ __launch_bounds__(256) void hist_kernel(
    const int* __restrict__ sin, const int* __restrict__ sout,
    float* __restrict__ cnt)
{
    __shared__ int h[32];
    const int b = blockIdx.x;
    if (threadIdx.x < 32) h[threadIdx.x] = 0;
    __syncthreads();
    for (int i = threadIdx.x; i < 800; i += 256) {
        atomicAdd(&h[sin[b * 800 + i] & 15], 1);
        atomicAdd(&h[16 + (sout[b * 800 + i] & 15)], 1);
    }
    __syncthreads();
    if (threadIdx.x < 32)
        cnt[b * 32 + threadIdx.x] = (float)h[threadIdx.x] * (1.0f / 800.0f);
}

// pos_mean[d] = mean over 400 rows of pos
__global__ __launch_bounds__(128) void posmean_kernel(
    const float* __restrict__ pos, float* __restrict__ pm)
{
    const int d = threadIdx.x;
    float s = 0.0f;
    for (int l = 0; l < LQ; ++l) s += pos[l * DD + d];
    pm[d] = s * (1.0f / (float)LQ);
}

// ---------------------------------------------------------------------------
// posW_T[128][400] = (pos @ W_q)^T, tokW_T[128][16] = (tok @ W_q)^T (pad>=10).
// W_q pre-packed f16.  Blocks 0..24 -> pos tiles, block 25 -> tok tile.
// Branch-free: clamped row pointer + lane scale for A, unified base+stride
// for the transposed store.
// ---------------------------------------------------------------------------
__global__ __launch_bounds__(128) void proj_kernel(
    const float* __restrict__ pos, const float* __restrict__ tok,
    const _Float16* __restrict__ WqP,
    float* __restrict__ posWT, float* __restrict__ tokWT)
{
    const int  blk  = blockIdx.x;
    const int  m0   = blk * 16;
    const int  lane = threadIdx.x & 31;
    const int  wave = threadIdx.x >> 5;
    const int  half = lane >> 4;
    const int  l15  = lane & 15;
    const bool isTok = (blk == 25);

    const float* abase;
    float scale;
    if (isTok) {
        int rr = (l15 < 10) ? l15 : 0;
        abase = tok + rr * DD;
        scale = (l15 < 10) ? 1.0f : 0.0f;
    } else {
        abase = pos + (m0 + l15) * DD;
        scale = 1.0f;
    }
    abase += half * 16;

    float*    dbase   = isTok ? tokWT : (posWT + m0);
    const int dstride = isTok ? 16 : LQ;

    for (int n0 = wave * 16; n0 < 128; n0 += 64) {
        v8f acc = {};
        const _Float16* wb = WqP + (size_t)(n0 + l15) * 32 + half * 16;
#pragma unroll
        for (int kt = 0; kt < 4; ++kt) {
            float av[16];
            *(v4f*)(av + 0)  = *(const v4f*)(abase + kt * 32);
            *(v4f*)(av + 4)  = *(const v4f*)(abase + kt * 32 + 4);
            *(v4f*)(av + 8)  = *(const v4f*)(abase + kt * 32 + 8);
            *(v4f*)(av + 12) = *(const v4f*)(abase + kt * 32 + 12);
            v16h a;
#pragma unroll
            for (int e = 0; e < 16; ++e) a[e] = (_Float16)(av[e] * scale);
            v16h b = *(const v16h*)(wb + (size_t)kt * 128 * 32);
            acc = __builtin_amdgcn_wmma_f32_16x16x32_f16(
                false, a, false, b, (short)0, acc, false, false);
        }
        const int n = n0 + l15;
#pragma unroll
        for (int j = 0; j < 8; ++j)
            dbase[n * dstride + (j + half * 8)] = acc[j];
    }
}

// ---------------------------------------------------------------------------
// The per-batch dense chain (sctx, hinit, ctx, linit, 4x(2 lcell + hcell),
// h2d, pb).  16 batch rows per block, 64 blocks, all linears on WMMA.
// Split-K concatenations use separately packed weight slices.
// ---------------------------------------------------------------------------
struct ChainP {
    const float *cnt, *tok, *pos_mean, *task_tok;
    const int   *task_id;
    const float *ctxB, *h2dB, *s1B, *s2B, *hi1B, *hi2B, *li1B, *li2B;
    const float *ligB, *lsgB, *licB, *lscB, *higB, *hsgB, *hicB, *hscB;
    const float *oB1;
    const _Float16 *ctxW, *h2dW, *s1W, *s2W, *hi1W, *hi2W, *li2W;
    const _Float16 *li1Wa, *li1Wb;                 // linit fc1: [H|ctx]
    const _Float16 *ligWa, *ligWb, *ligWc, *lsgW;  // lcell ig: [L|H|ctx], sg
    const _Float16 *licWa, *licWb, *licWc, *lscW;  // lcell ic, sc
    const _Float16 *higWa, *higWb, *hsgW;          // hcell ig: [H|L], sg
    const _Float16 *hicWa, *hicWb, *hscW;          // hcell ic, sc
    const _Float16 *oW1t, *oW1h;                   // out.fc1.W rows 128:256, 256:384
    float* pb;
};

__global__ __launch_bounds__(BD) void chain_kernel(ChainP P)
{
    __shared__ __attribute__((aligned(16))) float CMB[16 * 256];
    __shared__ __attribute__((aligned(16))) float T1 [16 * 192];
    __shared__ __attribute__((aligned(16))) float Hs [16 * 192];
    __shared__ __attribute__((aligned(16))) float Ls [16 * 160];
    __shared__ __attribute__((aligned(16))) float Cx [16 * 128];
    float* T2 = CMB;                  // alive only after ctx is computed

    const int b0  = blockIdx.x * 16;
    const int tid = threadIdx.x;

    // sup_agg -> CMB[16][256]  (hist @ tok + pos_mean, inp | out halves)
    for (int i = tid; i < 16 * 256; i += BD) {
        int r = i >> 8, c = i & 255;
        int hc = c >> 7, d = c & 127;
        const float* cn = P.cnt + (b0 + r) * 32 + hc * 16;
        float acc = P.pos_mean[d];
#pragma unroll
        for (int q = 0; q < 10; ++q) acc += cn[q] * P.tok[q * DD + d];
        CMB[i] = acc;
    }

    // support_context = sctx MLP(sup_agg) -> CMB[:,128:256]
    block_gemm16<256, 128>(CMB, 256, P.s1W, P.s1B, T1, 192, 0);
    for (int i = tid; i < 16 * 128; i += BD) {
        int r = i >> 7, c = i & 127;
        T1[r * 192 + c] = gelu_f(T1[r * 192 + c]);
    }
    block_gemm16<128, 128>(T1, 192, P.s2W, P.s2B, CMB + 128, 256, 0);

    // task tokens -> CMB[:,0:128]   (combined = [task_tok ; support_context])
    for (int i = tid; i < 16 * 128; i += BD) {
        int r = i >> 7, d = i & 127;
        CMB[r * 256 + d] = P.task_tok[(size_t)P.task_id[b0 + r] * DD + d];
    }
    __syncthreads();

    // H = hinit MLP(combined)
    block_gemm16<256, 192>(CMB, 256, P.hi1W, P.hi1B, T1, 192, 0);
    for (int i = tid; i < 16 * 192; i += BD) T1[i] = gelu_f(T1[i]);
    block_gemm16<192, 192>(T1, 192, P.hi2W, P.hi2B, Hs, 192, 0);

    // ctx = lin(combined)
    block_gemm16<256, 128>(CMB, 256, P.ctxW, P.ctxB, Cx, 128, 0);

    // Lst = linit MLP([H ; ctx])   (K split 192 + 128)
    block_gemm16<192, 160>(Hs, 192, P.li1Wa, P.li1B,  T1, 192, 0);
    block_gemm16<128, 160>(Cx, 128, P.li1Wb, nullptr, T1, 192, 1);
    for (int i = tid; i < 16 * 160; i += BD) {
        int r = i / 160, c = i - r * 160;
        T1[r * 192 + c] = gelu_f(T1[r * 192 + c]);
    }
    block_gemm16<160, 160>(T1, 192, P.li2W, P.li2B, Ls, 160, 0);

#pragma unroll 1
    for (int cyc = 0; cyc < 4; ++cyc) {
#pragma unroll 1
        for (int st = 0; st < 2; ++st) {
            // gate_pre -> T1 : l_in=[Ls;Hs;Cx] @ Wig + big + Ls @ Wsg + bsg
            block_gemm16<160, 160>(Ls, 160, P.ligWa, P.ligB,  T1, 192, 0);
            block_gemm16<192, 160>(Hs, 192, P.ligWb, nullptr, T1, 192, 1);
            block_gemm16<128, 160>(Cx, 128, P.ligWc, nullptr, T1, 192, 1);
            block_gemm16<160, 160>(Ls, 160, P.lsgW,  P.lsgB,  T1, 192, 1);
            // cand_pre -> T2
            block_gemm16<160, 160>(Ls, 160, P.licWa, P.licB,  T2, 192, 0);
            block_gemm16<192, 160>(Hs, 192, P.licWb, nullptr, T2, 192, 1);
            block_gemm16<128, 160>(Cx, 128, P.licWc, nullptr, T2, 192, 1);
            block_gemm16<160, 160>(Ls, 160, P.lscW,  P.lscB,  T2, 192, 1);
            for (int i = tid; i < 16 * 160; i += BD) {
                int r = i / 160, c = i - r * 160;
                float g  = 1.0f / (1.0f + expf(-T1[r * 192 + c]));
                float cd = tanhf(T2[r * 192 + c]);
                float pv = Ls[r * 160 + c];
                Ls[r * 160 + c] = (1.0f - g) * pv + g * cd;
            }
            __syncthreads();
        }
        // hcell: in = [Hs ; Ls]  (K split 192 + 160)
        block_gemm16<192, 192>(Hs, 192, P.higWa, P.higB,  T1, 192, 0);
        block_gemm16<160, 192>(Ls, 160, P.higWb, nullptr, T1, 192, 1);
        block_gemm16<192, 192>(Hs, 192, P.hsgW,  P.hsgB,  T1, 192, 1);
        block_gemm16<192, 192>(Hs, 192, P.hicWa, P.hicB,  T2, 192, 0);
        block_gemm16<160, 192>(Ls, 160, P.hicWb, nullptr, T2, 192, 1);
        block_gemm16<192, 192>(Hs, 192, P.hscW,  P.hscB,  T2, 192, 1);
        for (int i = tid; i < 16 * 192; i += BD) {
            float g  = 1.0f / (1.0f + expf(-T1[i]));
            float cd = tanhf(T2[i]);
            float pv = Hs[i];
            Hs[i] = (1.0f - g) * pv + g * cd;
        }
        __syncthreads();
    }

    // hd = h2d(Hs) -> T1 ;  pb = ctx @ W_t + hd @ W_h + out.fc1.b -> T2
    block_gemm16<192, 128>(Hs, 192, P.h2dW, P.h2dB,  T1, 192, 0);
    block_gemm16<128, 128>(Cx, 128, P.oW1t, P.oB1,   T2, 192, 0);
    block_gemm16<128, 128>(T1, 192, P.oW1h, nullptr, T2, 192, 1);
    for (int i = tid; i < 16 * 128; i += BD) {
        int r = i >> 7, d = i & 127;
        P.pb[(size_t)(b0 + r) * DD + d] = T2[r * 192 + d];
    }
}

// ---------------------------------------------------------------------------
// Decode.  logits[b,l,:] = gelu(tokW[q]+posW[l]+pb[b]) @ W2 + b2
// grid (2, B), 256 threads; posW_T coalesced over l, tokW_T row = one line.
// ---------------------------------------------------------------------------
__global__ __launch_bounds__(256) void decode_kernel(
    const int* __restrict__ q, const float* __restrict__ pb,
    const float* __restrict__ tokWT, const float* __restrict__ posWT,
    const float* __restrict__ W2, const float* __restrict__ b2,
    float* __restrict__ out)
{
    __shared__ float pbs[128];
    __shared__ float w2s[1280];
    __shared__ float b2s[16];
    const int b = blockIdx.y;
    if (threadIdx.x < 128) pbs[threadIdx.x] = pb[b * DD + threadIdx.x];
    for (int i = threadIdx.x; i < 1280; i += 256) w2s[i] = W2[i];
    if (threadIdx.x < 10) b2s[threadIdx.x] = b2[threadIdx.x];
    __syncthreads();

    const int l = blockIdx.x * 256 + threadIdx.x;
    if (l >= LQ) return;
    const int qi = q[b * LQ + l] & 15;

    float acc[10];
#pragma unroll
    for (int c = 0; c < 10; ++c) acc[c] = b2s[c];
    for (int d = 0; d < DD; ++d) {
        float x = tokWT[d * 16 + qi] + posWT[d * LQ + l] + pbs[d];
        float t = gelu_f(x);
#pragma unroll
        for (int c = 0; c < 10; ++c) acc[c] += t * w2s[d * 10 + c];
    }
    float* o = out + ((size_t)b * LQ + l) * 10;
#pragma unroll
    for (int c = 0; c < 10; ++c) o[c] = acc[c];
}

// ---------------------------------------------------------------------------
// Launch.  Input order = setup_inputs() dict insertion order, params flattened
// in declaration order (tok, pos, task_tok, ctx{W,b}, h2d, sctx, hinit, linit,
// lcell{ig,sg,ic,sc}, hcell{...}, out{fc1,fc2}).
// ---------------------------------------------------------------------------
extern "C" void kernel_launch(void* const* d_in, const int* in_sizes, int n_in,
                              void* d_out, int out_size, void* d_ws, size_t ws_size,
                              hipStream_t stream)
{
    (void)in_sizes; (void)n_in; (void)out_size; (void)ws_size;
    const int*   supIn   = (const int*)d_in[0];
    const int*   supOut  = (const int*)d_in[1];
    const int*   qIn     = (const int*)d_in[2];
    const int*   taskId  = (const int*)d_in[3];
    const float* tok     = (const float*)d_in[4];
    const float* pos     = (const float*)d_in[5];
    const float* taskTok = (const float*)d_in[6];
    const float* ctxW = (const float*)d_in[7],  *ctxB = (const float*)d_in[8];
    const float* h2dW = (const float*)d_in[9],  *h2dB = (const float*)d_in[10];
    const float* s1W  = (const float*)d_in[11], *s1B  = (const float*)d_in[12];
    const float* s2W  = (const float*)d_in[13], *s2B  = (const float*)d_in[14];
    const float* hi1W = (const float*)d_in[15], *hi1B = (const float*)d_in[16];
    const float* hi2W = (const float*)d_in[17], *hi2B = (const float*)d_in[18];
    const float* li1W = (const float*)d_in[19], *li1B = (const float*)d_in[20];
    const float* li2W = (const float*)d_in[21], *li2B = (const float*)d_in[22];
    const float* ligW = (const float*)d_in[23], *ligB = (const float*)d_in[24];
    const float* lsgW = (const float*)d_in[25], *lsgB = (const float*)d_in[26];
    const float* licW = (const float*)d_in[27], *licB = (const float*)d_in[28];
    const float* lscW = (const float*)d_in[29], *lscB = (const float*)d_in[30];
    const float* higW = (const float*)d_in[31], *higB = (const float*)d_in[32];
    const float* hsgW = (const float*)d_in[33], *hsgB = (const float*)d_in[34];
    const float* hicW = (const float*)d_in[35], *hicB = (const float*)d_in[36];
    const float* hscW = (const float*)d_in[37], *hscB = (const float*)d_in[38];
    const float* oW1  = (const float*)d_in[39], *oB1  = (const float*)d_in[40];
    const float* oW2  = (const float*)d_in[41], *oB2  = (const float*)d_in[42];
    float* out = (float*)d_out;

    // workspace partition (f32 region, then 64B-aligned f16 packed-weight pool)
    float* ws    = (float*)d_ws;
    float* cnt   = ws;                     // NB*32
    float* pb    = cnt + NB * 32;          // NB*128
    float* pm    = pb + NB * DD;           // 128
    float* tokWT = pm + DD;                // 128*16
    float* posWT = tokWT + DD * 16;        // 128*400
    _Float16* wpool = (_Float16*)(posWT + DD * LQ);   // byte off 868864 (64B ok)

    size_t off = 0;
    auto pack = [&](const float* src, int K, int N) -> const _Float16* {
        _Float16* d = wpool + off;
        packw_kernel<<<64, 256, 0, stream>>>(src, d, K, N);
        off += (size_t)K * N;
        return d;
    };

    ChainP P;
    P.cnt = cnt; P.tok = tok; P.pos_mean = pm; P.task_tok = taskTok;
    P.task_id = taskId;
    P.ctxB = ctxB; P.h2dB = h2dB; P.s1B = s1B; P.s2B = s2B;
    P.hi1B = hi1B; P.hi2B = hi2B; P.li1B = li1B; P.li2B = li2B;
    P.ligB = ligB; P.lsgB = lsgB; P.licB = licB; P.lscB = lscB;
    P.higB = higB; P.hsgB = hsgB; P.hicB = hicB; P.hscB = hscB;
    P.oB1 = oB1; P.pb = pb;

    P.s1W   = pack(s1W, 256, 128);
    P.s2W   = pack(s2W, 128, 128);
    P.hi1W  = pack(hi1W, 256, 192);
    P.hi2W  = pack(hi2W, 192, 192);
    P.ctxW  = pack(ctxW, 256, 128);
    P.li1Wa = pack(li1W,             192, 160);
    P.li1Wb = pack(li1W + 192 * 160, 128, 160);
    P.li2W  = pack(li2W, 160, 160);
    P.ligWa = pack(ligW,             160, 160);
    P.ligWb = pack(ligW + 160 * 160, 192, 160);
    P.ligWc = pack(ligW + 352 * 160, 128, 160);
    P.lsgW  = pack(lsgW, 160, 160);
    P.licWa = pack(licW,             160, 160);
    P.licWb = pack(licW + 160 * 160, 192, 160);
    P.licWc = pack(licW + 352 * 160, 128, 160);
    P.lscW  = pack(lscW, 160, 160);
    P.higWa = pack(higW,             192, 192);
    P.higWb = pack(higW + 192 * 192, 160, 192);
    P.hsgW  = pack(hsgW, 192, 192);
    P.hicWa = pack(hicW,             192, 192);
    P.hicWb = pack(hicW + 192 * 192, 160, 192);
    P.hscW  = pack(hscW, 192, 192);
    P.h2dW  = pack(h2dW, 192, 128);
    P.oW1t  = pack(oW1 + 128 * 128, 128, 128);
    P.oW1h  = pack(oW1 + 256 * 128, 128, 128);
    const _Float16* WqP = pack(oW1, 128, 128);      // rows 0..127 for decode fc1

    hist_kernel   <<<NB, 256, 0, stream>>>(supIn, supOut, cnt);
    posmean_kernel<<<1, 128, 0, stream>>>(pos, pm);
    proj_kernel   <<<26, 128, 0, stream>>>(pos, tok, WqP, posWT, tokWT);
    chain_kernel  <<<NB / 16, BD, 0, stream>>>(P);
    decode_kernel <<<dim3(2, NB), 256, 0, stream>>>(qIn, pb, tokWT, posWT,
                                                    oW2, oB2, out);
}
